// CounterPropagationNetwork_85650237817449
// MI455X (gfx1250) — matrix operand order repaired and checked by reading, not compiled, not run on hardware
//
#include <hip/hip_runtime.h>
#include <cstdint>
#include <cstddef>

// Problem sizes (fixed by the reference)
#define B_N    16384
#define D_IN   512
#define H_N    4096
#define D_OUT_ 1024

// GEMM tiling
#define BM 64      // rows per block
#define BN 128     // columns per N-iteration (whole H swept per block)
#define KSTEP 32   // K per wmma_f32_16x16x32_bf16

typedef __bf16 bf16t;
typedef bf16t  v16bf __attribute__((ext_vector_type(16)));
typedef float  v8f   __attribute__((ext_vector_type(8)));

struct B32x2 { uint4 lo; uint4 hi; };

static __device__ __forceinline__ v16bf frag_from(uint4 a, uint4 b) {
  B32x2 t; t.lo = a; t.hi = b;
  return __builtin_bit_cast(v16bf, t);
}

// round-to-nearest-even f32 -> bf16 (as raw u16)
static __device__ __forceinline__ unsigned short f2bf(float f) {
  unsigned u = __float_as_uint(f);
  u += 0x7FFFu + ((u >> 16) & 1u);
  return (unsigned short)(u >> 16);
}
static __device__ __forceinline__ float bf2f(unsigned short s) {
  return __uint_as_float((unsigned)s << 16);
}

static __device__ __forceinline__ v8f wmma_bf16(v16bf a, v16bf b, v8f c) {
  return __builtin_amdgcn_wmma_f32_16x16x32_bf16(false, a, false, b, (short)0, c,
                                                 false, false);
}

// A fragment (16x32 bf16): lane L holds row L%16; k-chunks at 8*(L/16) and 16+8*(L/16)
static __device__ __forceinline__ v16bf load_a(const unsigned short* rowp, int k, int lhi) {
  const uint4* p0 = reinterpret_cast<const uint4*>(rowp + k + lhi * 8);
  const uint4* p1 = reinterpret_cast<const uint4*>(rowp + k + lhi * 8 + 16);
  return frag_from(*p0, *p1);
}
// B fragment (32x16 bf16): lane L holds K-row (column) L%16; k-chunks at 8*(L/16) and 16+8*(L/16)
static __device__ __forceinline__ v16bf load_b(const unsigned short* colp, int k, int lhi) {
  const uint4* p0 = reinterpret_cast<const uint4*>(colp + k + lhi * 8);
  const uint4* p1 = reinterpret_cast<const uint4*>(colp + k + lhi * 8 + 16);
  return frag_from(*p0, *p1);
}

// ---------------- prep kernels ----------------

__global__ void convert_split_kernel(const float* __restrict__ src,
                                     unsigned short* __restrict__ hi,
                                     unsigned short* __restrict__ lo, int n) {
  int i = blockIdx.x * blockDim.x + threadIdx.x;
  if (i < n) {
    float v = src[i];
    unsigned short h = f2bf(v);
    float r = v - bf2f(h);
    hi[i] = h;
    lo[i] = f2bf(r);
  }
}

// one wave32 per Kohonen row: kn[h] = 0.5 * ||k_h||^2  (full f32)
__global__ void knorm_kernel(const float* __restrict__ K, float* __restrict__ kn) {
  int gwave = (blockIdx.x * blockDim.x + threadIdx.x) >> 5;
  int lane = threadIdx.x & 31;
  if (gwave < H_N) {
    const float* row = K + (size_t)gwave * D_IN;
    float s = 0.f;
    for (int k = lane; k < D_IN; k += 32) { float v = row[k]; s += v * v; }
#pragma unroll
    for (int off = 16; off > 0; off >>= 1) s += __shfl_down(s, off, 32);
    if (lane == 0) kn[gwave] = 0.5f * s;
  }
}

// Gt[h, d] = G[d, h]   (so the gather reads coalesced rows)
__global__ void transpose_g_kernel(const float* __restrict__ G, float* __restrict__ Gt) {
  __shared__ float tile[32][33];
  const int hb = blockIdx.x * 32;
  const int db = blockIdx.y * 32;
  const int tx = threadIdx.x, ty = threadIdx.y;  // blockDim (32, 8)
#pragma unroll
  for (int j = 0; j < 32; j += 8)
    tile[ty + j][tx] = G[(size_t)(db + ty + j) * H_N + hb + tx];
  __syncthreads();
#pragma unroll
  for (int j = 0; j < 32; j += 8)
    Gt[(size_t)(hb + ty + j) * D_OUT_ + db + tx] = tile[tx][ty + j];
}

// ---------------- fused score GEMM + per-row argmax ----------------

__global__ __launch_bounds__(256) void cpn_score_argmax_kernel(
    const unsigned short* __restrict__ Xhi, const unsigned short* __restrict__ Xlo,
    const unsigned short* __restrict__ Khi, const unsigned short* __restrict__ Klo,
    const float* __restrict__ kn, int* __restrict__ winners) {
  const int tid   = threadIdx.x;
  const int lane  = tid & 31;
  const int wave  = tid >> 5;   // 0..7
  const int waveM = wave & 1;   // 2 waves along M (32 rows each)
  const int waveN = wave >> 1;  // 4 waves along N (32 cols each)
  const int lmod  = lane & 15;
  const int lhi   = lane >> 4;

  const int rowBlock = blockIdx.x * BM;

  // per-lane A row pointers (A layout: lane holds row lane%16)
  const unsigned short* a0h = Xhi + (size_t)(rowBlock + waveM * 32 + 0 + lmod) * D_IN;
  const unsigned short* a1h = Xhi + (size_t)(rowBlock + waveM * 32 + 16 + lmod) * D_IN;
  const unsigned short* a0l = Xlo + (size_t)(rowBlock + waveM * 32 + 0 + lmod) * D_IN;
  const unsigned short* a1l = Xlo + (size_t)(rowBlock + waveM * 32 + 16 + lmod) * D_IN;

  float bestVal[2][8];
  int   bestIdx[2][8];
#pragma unroll
  for (int ms = 0; ms < 2; ++ms)
#pragma unroll
    for (int i = 0; i < 8; ++i) { bestVal[ms][i] = -3.402823466e38f; bestIdx[ms][i] = 0; }

  for (int it = 0; it < H_N / BN; ++it) {
    const int nbase = it * BN + waveN * 32;
    const int col0 = nbase + lmod;        // nsub 0 column for this lane
    const int col1 = nbase + 16 + lmod;   // nsub 1 column for this lane
    const unsigned short* b0h = Khi + (size_t)col0 * D_IN;
    const unsigned short* b1h = Khi + (size_t)col1 * D_IN;
    const unsigned short* b0l = Klo + (size_t)col0 * D_IN;
    const unsigned short* b1l = Klo + (size_t)col1 * D_IN;
    const float bias0 = -kn[col0];  // fold -0.5*||k||^2 into the accumulator (f32 exact)
    const float bias1 = -kn[col1];

    v8f acc00, acc01, acc10, acc11;
#pragma unroll
    for (int i = 0; i < 8; ++i) {
      acc00[i] = bias0; acc01[i] = bias1;
      acc10[i] = bias0; acc11[i] = bias1;
    }

    for (int k = 0; k < D_IN; k += KSTEP) {
      v16bf fa0h = load_a(a0h, k, lhi);
      v16bf fa1h = load_a(a1h, k, lhi);
      v16bf fa0l = load_a(a0l, k, lhi);
      v16bf fa1l = load_a(a1l, k, lhi);
      v16bf fb0h = load_b(b0h, k, lhi);
      v16bf fb1h = load_b(b1h, k, lhi);
      v16bf fb0l = load_b(b0l, k, lhi);
      v16bf fb1l = load_b(b1l, k, lhi);

      // compensated product: xhi*khi + xhi*klo + xlo*khi (error ~ xlo*klo ~ 1e-5 rel)
      acc00 = wmma_bf16(fa0h, fb0h, acc00);
      acc01 = wmma_bf16(fa0h, fb1h, acc01);
      acc10 = wmma_bf16(fa1h, fb0h, acc10);
      acc11 = wmma_bf16(fa1h, fb1h, acc11);

      acc00 = wmma_bf16(fa0h, fb0l, acc00);
      acc01 = wmma_bf16(fa0h, fb1l, acc01);
      acc10 = wmma_bf16(fa1h, fb0l, acc10);
      acc11 = wmma_bf16(fa1h, fb1l, acc11);

      acc00 = wmma_bf16(fa0l, fb0h, acc00);
      acc01 = wmma_bf16(fa0l, fb1h, acc01);
      acc10 = wmma_bf16(fa1l, fb0h, acc10);
      acc11 = wmma_bf16(fa1l, fb1h, acc11);
    }

    // running per-row-slot argmax (n ascending within a lane -> strict > keeps first)
#pragma unroll
    for (int i = 0; i < 8; ++i) {
      if (acc00[i] > bestVal[0][i]) { bestVal[0][i] = acc00[i]; bestIdx[0][i] = col0; }
      if (acc01[i] > bestVal[0][i]) { bestVal[0][i] = acc01[i]; bestIdx[0][i] = col1; }
      if (acc10[i] > bestVal[1][i]) { bestVal[1][i] = acc10[i]; bestIdx[1][i] = col0; }
      if (acc11[i] > bestVal[1][i]) { bestVal[1][i] = acc11[i]; bestIdx[1][i] = col1; }
    }
  }

  // C/D layout: vgpr i, lane L -> row = i + 8*(L/16), col = L%16.
  // Reduce across the 16 lanes (same half) that share a row, with lowest-index tiebreak.
  __shared__ float sval[4][BM];
  __shared__ int   sidx[4][BM];
#pragma unroll
  for (int ms = 0; ms < 2; ++ms) {
#pragma unroll
    for (int i = 0; i < 8; ++i) {
      float v = bestVal[ms][i];
      int   n = bestIdx[ms][i];
#pragma unroll
      for (int off = 1; off < 16; off <<= 1) {
        float ov = __shfl_xor(v, off, 32);
        int   on = __shfl_xor(n, off, 32);
        if (ov > v || (ov == v && on < n)) { v = ov; n = on; }
      }
      if (lmod == 0) {
        int r = waveM * 32 + ms * 16 + i + 8 * lhi;  // 0..63
        sval[waveN][r] = v;
        sidx[waveN][r] = n;
      }
    }
  }
  __syncthreads();

  if (tid < BM) {
    float v = sval[0][tid];
    int   n = sidx[0][tid];
#pragma unroll
    for (int w = 1; w < 4; ++w) {
      float ov = sval[w][tid];
      int   on = sidx[w][tid];
      if (ov > v || (ov == v && on < n)) { v = ov; n = on; }
    }
    winners[rowBlock + tid] = n;
  }
}

// ---------------- output gather ----------------

__global__ void gather_out_kernel(const float* __restrict__ Gt,
                                  const int* __restrict__ winners,
                                  float* __restrict__ out) {
  const int row = blockIdx.x;
  const int w = winners[row];
  const float* src = Gt + (size_t)w * D_OUT_;
  float* dst = out + (size_t)row * D_OUT_;
  for (int d = threadIdx.x; d < D_OUT_; d += blockDim.x) dst[d] = src[d];
}

// ---------------- host launch ----------------

extern "C" void kernel_launch(void* const* d_in, const int* in_sizes, int n_in,
                              void* d_out, int out_size, void* d_ws, size_t ws_size,
                              hipStream_t stream) {
  (void)in_sizes; (void)n_in; (void)out_size; (void)ws_size;
  const float* x = (const float*)d_in[0];                 // [B, D_IN]
  const float* K = (const float*)d_in[1];                 // [H, D_IN]
  const float* G = (const float*)d_in[2];                 // [D_OUT, H]

  // workspace layout (~56 MB)
  char* ws = (char*)d_ws;
  size_t off = 0;
  unsigned short* Xhi = (unsigned short*)(ws + off); off += (size_t)B_N * D_IN * 2;
  unsigned short* Xlo = (unsigned short*)(ws + off); off += (size_t)B_N * D_IN * 2;
  unsigned short* Khi = (unsigned short*)(ws + off); off += (size_t)H_N * D_IN * 2;
  unsigned short* Klo = (unsigned short*)(ws + off); off += (size_t)H_N * D_IN * 2;
  float* kn = (float*)(ws + off);                    off += (size_t)H_N * 4;
  float* Gt = (float*)(ws + off);                    off += (size_t)H_N * D_OUT_ * 4;

  float* out = (float*)d_out;                                    // [B, D_OUT]
  int* winners = (int*)d_out + (size_t)B_N * D_OUT_;             // [B] int32

  {
    int n = B_N * D_IN;
    convert_split_kernel<<<(n + 255) / 256, 256, 0, stream>>>(x, Xhi, Xlo, n);
  }
  {
    int n = H_N * D_IN;
    convert_split_kernel<<<(n + 255) / 256, 256, 0, stream>>>(K, Khi, Klo, n);
  }
  knorm_kernel<<<H_N / 8, 256, 0, stream>>>(K, kn);
  transpose_g_kernel<<<dim3(H_N / 32, D_OUT_ / 32), dim3(32, 8), 0, stream>>>(G, Gt);
  cpn_score_argmax_kernel<<<B_N / BM, 256, 0, stream>>>(Xhi, Xlo, Khi, Klo, kn, winners);
  gather_out_kernel<<<B_N, 256, 0, stream>>>(Gt, winners, out);
}
